// SemanticDecoupling_24369644437981
// MI455X (gfx1250) — compile-verified, matrix operand order checked by reading, not run on hardware
//
#include <hip/hip_runtime.h>
#include <math.h>

typedef __attribute__((ext_vector_type(16))) __bf16 v16bf;
typedef __attribute__((ext_vector_type(8)))  __bf16 v8bf;
typedef __attribute__((ext_vector_type(8)))  float  v8f;

constexpr int kB = 8, kS = 14, kD = 2048, kC = 20, kWD = 300, kI = 1024;
constexpr int kP = kS * kS;          // 196 spatial positions
constexpr int kN = kB * kP;          // 1568 rows

// ---------------------------------------------------------------- fp32 -> bf16
__global__ void cvt_bf16_kernel(const float* __restrict__ in, __bf16* __restrict__ out, int n) {
    int i = blockIdx.x * blockDim.x + threadIdx.x;
    if (i < n) out[i] = (__bf16)in[i];
}

// ---------------------------------------------------- h_img = Xbf16 @ W1bf16^T
// A: [kN, kD] bf16 row-major (K contiguous), Bm: [kI, kD] bf16 (K contiguous).
// One wave per 16x16 output tile; K loop in steps of 32 with v_wmma_f32_16x16x32_bf16.
__global__ void wmma_himg_gemm(const __bf16* __restrict__ A,
                               const __bf16* __restrict__ Bm,
                               float* __restrict__ C) {
    const int wave = (blockIdx.x * blockDim.x + threadIdx.x) >> 5;
    const int lane = threadIdx.x & 31;
    const int tilesN = kI / 16;                 // 64
    const int tm = wave / tilesN;               // 0..97
    const int tn = wave % tilesN;
    if (tm >= kN / 16) return;
    const int half = lane >> 4;                 // K-half select (ISA 16-bit A/B layout)
    const int r16  = lane & 15;                 // M (A) / N (B) index within tile
    const __bf16* Arow = A + (size_t)(tm * 16 + r16) * kD;
    const __bf16* Brow = Bm + (size_t)(tn * 16 + r16) * kD;
    v8f acc = {0.f, 0.f, 0.f, 0.f, 0.f, 0.f, 0.f, 0.f};
    for (int k0 = 0; k0 < kD; k0 += 32) {
        __builtin_prefetch(Arow + k0 + 256, 0, 3);   // global_prefetch_b8 (near-cache)
        __builtin_prefetch(Brow + k0 + 256, 0, 3);
        // Lane holds K = k0 + {0..7,16..23} (half=0) or {8..15,24..31} (half=1):
        // two contiguous 16B loads per fragment.
        v8bf alo = *(const v8bf*)(Arow + k0 + half * 8);
        v8bf ahi = *(const v8bf*)(Arow + k0 + 16 + half * 8);
        v8bf blo = *(const v8bf*)(Brow + k0 + half * 8);
        v8bf bhi = *(const v8bf*)(Brow + k0 + 16 + half * 8);
        v16bf a, bf;
#pragma unroll
        for (int e = 0; e < 8; ++e) {
            a[e] = alo[e];  a[e + 8] = ahi[e];
            bf[e] = blo[e]; bf[e + 8] = bhi[e];
        }
        acc = __builtin_amdgcn_wmma_f32_16x16x32_bf16(
            /*neg_a=*/false, a, /*neg_b=*/false, bf,
            /*c_mod=*/(short)0, acc, /*reuse_a=*/false, /*reuse_b=*/false);
    }
    // f32 C/D layout: lane 0-15 -> N=lane, M=r; lane 16-31 -> N=lane-16, M=r+8.
    const int n = tn * 16 + r16;
    const int mbase = tm * 16 + half * 8;
#pragma unroll
    for (int r = 0; r < 8; ++r)
        C[(size_t)(mbase + r) * kI + n] = acc[r];
}

// -------------------------------------------------------- h_word = word @ W2^T
__global__ void hword_kernel(const float* __restrict__ word, const float* __restrict__ W2,
                             float* __restrict__ hword) {
    int idx = blockIdx.x * blockDim.x + threadIdx.x;
    if (idx >= kC * kI) return;
    int c = idx / kI, i = idx % kI;
    float s = 0.f;
    for (int k = 0; k < kWD; ++k) s += word[c * kWD + k] * W2[i * kWD + k];
    hword[idx] = s;
}

// --------------------------- v = W4 @ W3 (length kI);  c0 = W4.b3 + b4  -------
__global__ void v_kernel(const float* __restrict__ W3, const float* __restrict__ W4,
                         const float* __restrict__ b3, const float* __restrict__ b4,
                         float* __restrict__ v, float* __restrict__ c0) {
    int k = blockIdx.x * blockDim.x + threadIdx.x;
    if (k < kI) {
        float s = 0.f;
        for (int j = 0; j < kI; ++j) s += W4[j] * W3[(size_t)j * kI + k];  // coalesced in k
        v[k] = s;
    }
    if (blockIdx.x == 0 && threadIdx.x == 0) {
        float s = 0.f;
        for (int j = 0; j < kI; ++j) s += W4[j] * b3[j];
        *c0 = s + b4[0];
    }
}

// ---- coef_raw[n,c] = c0 + sum_k tanh(h_img[n,k]*h_word[c,k]) * v[k] ----------
// One wave32 per (n,c); shuffle tree reduction.
__global__ void coef_kernel(const float* __restrict__ himg, const float* __restrict__ hword,
                            const float* __restrict__ v, const float* __restrict__ c0p,
                            float* __restrict__ coef_raw) {
    int gid = blockIdx.x * blockDim.x + threadIdx.x;
    int wave = gid >> 5, lane = gid & 31;
    if (wave >= kN * kC) return;
    int n = wave / kC, c = wave % kC;
    const float* hi = himg + (size_t)n * kI;
    const float* hw = hword + (size_t)c * kI;
    float s = 0.f;
    for (int k = lane; k < kI; k += 32)
        s += tanhf(hi[k] * hw[k]) * v[k];
#pragma unroll
    for (int off = 16; off > 0; off >>= 1) s += __shfl_down(s, off, 32);
    if (lane == 0) coef_raw[n * kC + c] = s + *c0p;
}

// ---------------- softmax over the 196 spatial positions per (b, c) -----------
__global__ void softmax_kernel(const float* __restrict__ raw,
                               float* __restrict__ coefn, float* __restrict__ coef_out) {
    int wave = (blockIdx.x * blockDim.x + threadIdx.x) >> 5;
    int lane = threadIdx.x & 31;
    if (wave >= kB * kC) return;
    int b = wave / kC, c = wave % kC;
    float mx = -3.0e38f;
    for (int p = lane; p < kP; p += 32) mx = fmaxf(mx, raw[(b * kP + p) * kC + c]);
#pragma unroll
    for (int off = 16; off > 0; off >>= 1) mx = fmaxf(mx, __shfl_xor(mx, off, 32));
    float sum = 0.f;
    for (int p = lane; p < kP; p += 32) sum += __expf(raw[(b * kP + p) * kC + c] - mx);
#pragma unroll
    for (int off = 16; off > 0; off >>= 1) sum += __shfl_xor(sum, off, 32);
    float inv = 1.0f / sum;
    for (int p = lane; p < kP; p += 32) {
        float e = __expf(raw[(b * kP + p) * kC + c] - mx) * inv;
        coefn[(b * kP + p) * kC + c] = e;
        coef_out[(b * kP + p) * kC + c] = e;
    }
}

__global__ void zero_kernel(float* __restrict__ p, int n) {
    int i = blockIdx.x * blockDim.x + threadIdx.x;
    if (i < n) p[i] = 0.f;
}

// ---- fmwc[b,y,x,c,d] = img*coef ; semantic[b,c,d] += fmwc (atomic) -----------
// Image element loaded once (NT) and fanned out to all 20 classes; the 257 MB
// fmwc stream uses non-temporal stores so it doesn't thrash the 192 MB L2.
__global__ void fmwc_kernel(const float* __restrict__ img, const float* __restrict__ coefn,
                            float* __restrict__ fmwc, float* __restrict__ semantic) {
    int blk = blockIdx.x;             // 256 blocks: b(8) x dchunk(8) x pgroup(4)
    int pg = blk & 3;
    int dchunk = (blk >> 2) & 7;
    int b = blk >> 5;
    int d = dchunk * 256 + threadIdx.x;
    float acc[kC];
#pragma unroll
    for (int c = 0; c < kC; ++c) acc[c] = 0.f;
    int p0 = pg * 49;
    for (int p = p0; p < p0 + 49; ++p) {
        float val = __builtin_nontemporal_load(&img[((size_t)(b * kP + p)) * kD + d]);
        const float* co = coefn + (b * kP + p) * kC;
#pragma unroll
        for (int c = 0; c < kC; ++c) {
            float f = val * co[c];
            __builtin_nontemporal_store(f, &fmwc[((size_t)((b * kP + p) * kC + c)) * kD + d]);
            acc[c] += f;
        }
    }
#pragma unroll
    for (int c = 0; c < kC; ++c)
        atomicAdd(&semantic[((size_t)(b * kC + c)) * kD + d], acc[c]);
}

extern "C" void kernel_launch(void* const* d_in, const int* in_sizes, int n_in,
                              void* d_out, int out_size, void* d_ws, size_t ws_size,
                              hipStream_t stream) {
    const float* img  = (const float*)d_in[0];   // [8,14,14,2048]
    const float* word = (const float*)d_in[1];   // [20,300]
    const float* W1   = (const float*)d_in[2];   // [1024,2048]
    const float* W2   = (const float*)d_in[3];   // [1024,300]
    const float* W3   = (const float*)d_in[4];   // [1024,1024]
    const float* b3   = (const float*)d_in[5];   // [1024]
    const float* W4   = (const float*)d_in[6];   // [1,1024]
    const float* b4   = (const float*)d_in[7];   // [1]

    float* out       = (float*)d_out;
    float* semantic  = out;                                          // 8*20*2048
    float* fmwc      = out + (size_t)kB * kC * kD;                   // +327680
    float* coef_out  = fmwc + (size_t)kN * kC * kD;                  // +64225280

    // ---- workspace layout (256B aligned slabs) ----
    char* ws = (char*)d_ws;
    size_t off = 0;
    auto take = [&](size_t bytes) { size_t o = off; off += (bytes + 255) & ~(size_t)255; return o; };
    __bf16* Xbf  = (__bf16*)(ws + take((size_t)kN * kD * sizeof(__bf16)));
    __bf16* W1bf = (__bf16*)(ws + take((size_t)kI * kD * sizeof(__bf16)));
    float* himg  = (float*)(ws + take((size_t)kN * kI * sizeof(float)));
    float* hword = (float*)(ws + take((size_t)kC * kI * sizeof(float)));
    float* vvec  = (float*)(ws + take((size_t)kI * sizeof(float)));
    float* c0    = (float*)(ws + take(256));
    float* craw  = (float*)(ws + take((size_t)kN * kC * sizeof(float)));
    float* cnorm = (float*)(ws + take((size_t)kN * kC * sizeof(float)));

    // 1) fp32 -> bf16 for the WMMA GEMM operands
    {
        int n1 = kN * kD;  // 3,211,264
        cvt_bf16_kernel<<<(n1 + 255) / 256, 256, 0, stream>>>(img, Xbf, n1);
        int n2 = kI * kD;  // 2,097,152
        cvt_bf16_kernel<<<(n2 + 255) / 256, 256, 0, stream>>>(W1, W1bf, n2);
    }
    // 2) h_img = X @ W1^T via v_wmma_f32_16x16x32_bf16 (98x64 tiles, 1 wave each)
    {
        int waves = (kN / 16) * (kI / 16);          // 6272
        wmma_himg_gemm<<<waves / 8, 256, 0, stream>>>(Xbf, W1bf, himg);
    }
    // 3) h_word = word @ W2^T
    hword_kernel<<<(kC * kI + 255) / 256, 256, 0, stream>>>(word, W2, hword);
    // 4) v = W4 @ W3, c0 = W4.b3 + b4  (fc3/fc4 collapsed: feat is never an output)
    v_kernel<<<(kI + 255) / 256, 256, 0, stream>>>(W3, W4, b3, b4, vvec, c0);
    // 5) coef_raw[n,c] = c0 + sum_k tanh(h_img*h_word)*v   (one wave per (n,c))
    {
        long threads = (long)kN * kC * 32;          // 1,003,520
        coef_kernel<<<(int)(threads / 256), 256, 0, stream>>>(himg, hword, vvec, c0, craw);
    }
    // 6) softmax over spatial positions per (b,c)
    {
        int threads = kB * kC * 32;                 // 5120
        softmax_kernel<<<threads / 256, 256, 0, stream>>>(craw, cnorm, coef_out);
    }
    // 7) zero semantic accumulator, then fused fmwc + atomic pooled sum
    zero_kernel<<<(kB * kC * kD + 255) / 256, 256, 0, stream>>>(semantic, kB * kC * kD);
    fmwc_kernel<<<kB * 8 * 4, 256, 0, stream>>>(img, cnorm, fmwc, semantic);
}
